// TopKAttention_88313117540987
// MI455X (gfx1250) — matrix operand order repaired and checked by reading, not compiled, not run on hardware
//
#include <hip/hip_runtime.h>
#include <math.h>

typedef __bf16 bf16;
typedef __attribute__((ext_vector_type(16))) __bf16 v16bf;
typedef __attribute__((ext_vector_type(8)))  __bf16 v8bf;
typedef __attribute__((ext_vector_type(8)))  float   v8f;

union Frag16 { v16bf v; v8bf h[2]; bf16 e[16]; };

static __device__ __forceinline__ float bf2f(bf16 x) {
  unsigned short s; __builtin_memcpy(&s, &x, 2);
  union { unsigned u; float f; } c; c.u = ((unsigned)s) << 16; return c.f;
}
// RNE truncate (finite inputs; no NaN guard needed)
static __device__ __forceinline__ bf16 f2bf(float f) {
  union { float f; unsigned u; } c; c.f = f;
  unsigned r = c.u + 0x7FFFu + ((c.u >> 16) & 1u);
  unsigned short s = (unsigned short)(r >> 16);
  bf16 out; __builtin_memcpy(&out, &s, 2); return out;
}

// ---------------------------------------------------------------------------
// f32 -> bf16 elementwise convert (pure bandwidth pass)
// ---------------------------------------------------------------------------
__global__ void cvt_f32_bf16_kernel(const float* __restrict__ in,
                                    bf16* __restrict__ out, int n) {
  int i = (blockIdx.x * blockDim.x + threadIdx.x) * 4;
  if (i + 3 < n) {
    float4 f = *(const float4*)(in + i);
    out[i+0] = f2bf(f.x); out[i+1] = f2bf(f.y);
    out[i+2] = f2bf(f.z); out[i+3] = f2bf(f.w);
  }
}

// ---------------------------------------------------------------------------
// WMMA GEMM:  C[m,n] = sum_k A[m,k] * W[n,k]      (C = A * W^T), all bf16 in.
// Each wave computes a (MT*16) x 64 tile of C: MT A-fragments, each B
// fragment reused MT times -> 12 VMEM per 8 WMMA at MT=2.
// 8 waves / 256-thread block.
// EPI: 0 = store bf16, 1 = store f32, 2 = scores (scale + causal mask, f32)
// ---------------------------------------------------------------------------
template <int EPI, int MT>
__global__ void wmma_gemm_kernel(const bf16* __restrict__ A, int lda,
                                 const bf16* __restrict__ W, int ldw,
                                 void* __restrict__ Cp, int ldc,
                                 int M, int N, int K, float scale) {
  const int lane  = threadIdx.x & 31;
  const int wave  = threadIdx.x >> 5;
  const int tilesN = N >> 6;                      // 64-wide N tiles
  const int wid   = blockIdx.x * (blockDim.x >> 5) + wave;
  const int tM    = wid / tilesN;
  const int tN    = wid % tilesN;
  if (tM * 16 * MT >= M) return;                  // wave-uniform

  const int aoff = (lane >> 4) * 8;               // A: K sub-offset 0 / 8
  const int boff = (lane >> 4) * 8;               // B: K sub-offset 0 / 8
  const bf16* Arow[MT];
#pragma unroll
  for (int im = 0; im < MT; ++im)
    Arow[im] = A + (size_t)(tM * 16 * MT + im * 16 + (lane & 15)) * (size_t)lda;
  const bf16* Wrow0 = W + (size_t)(tN * 64 + (lane & 15)) * (size_t)ldw;

  const v8f vzero = {0.f,0.f,0.f,0.f,0.f,0.f,0.f,0.f};
  v8f acc[MT * 4];
#pragma unroll
  for (int t = 0; t < MT * 4; ++t) acc[t] = vzero;

#pragma unroll 2
  for (int k0 = 0; k0 < K; k0 += 32) {
    Frag16 a[MT];
#pragma unroll
    for (int im = 0; im < MT; ++im) {
      a[im].h[0] = *(const v8bf*)(Arow[im] + k0 + aoff);
      a[im].h[1] = *(const v8bf*)(Arow[im] + k0 + 16 + aoff);
    }
    const int kb = k0 + boff;
#pragma unroll
    for (int t = 0; t < 4; ++t) {
      const bf16* wrow = Wrow0 + (size_t)(t * 16) * (size_t)ldw + kb;
      Frag16 b;
      b.h[0] = *(const v8bf*)(wrow);
      b.h[1] = *(const v8bf*)(wrow + 16);
#pragma unroll
      for (int im = 0; im < MT; ++im)
        acc[im * 4 + t] = __builtin_amdgcn_wmma_f32_16x16x32_bf16(
            false, a[im].v, false, b.v, (short)0, acc[im * 4 + t], false, false);
    }
  }

#pragma unroll
  for (int im = 0; im < MT; ++im) {
    const int row0 = tM * 16 * MT + im * 16 + (lane >> 4) * 8;
#pragma unroll
    for (int t = 0; t < 4; ++t) {
      const int col = tN * 64 + t * 16 + (lane & 15);
#pragma unroll
      for (int r = 0; r < 8; ++r) {
        float vv = acc[im * 4 + t][r];
        const int orow = row0 + r;
        const size_t oidx = (size_t)orow * (size_t)ldc + col;
        if (EPI == 2) {               // attention scores: scale + causal mask
          vv *= scale;
          if (col > orow) vv += -1.0e9f;
          ((float*)Cp)[oidx] = vv;
        } else if (EPI == 1) {
          ((float*)Cp)[oidx] = vv;
        } else {
          ((bf16*)Cp)[oidx] = f2bf(vv);
        }
      }
    }
  }
}

// ---------------------------------------------------------------------------
// RoPE in place on bf16 Q and K.  One thread per (pos, head, d<64) pair.
// ---------------------------------------------------------------------------
__global__ void rope_kernel(bf16* __restrict__ Q, bf16* __restrict__ Kk,
                            const int* __restrict__ pos_ids, int S) {
  int idx = blockIdx.x * blockDim.x + threadIdx.x;   // S*32*64
  int d = idx & 63;
  int t = idx >> 6;
  int h = t & 31;
  int i = t >> 5;
  if (i >= S) return;
  float p   = (float)pos_ids[i];
  float inv = __expf(-((2.0f * (float)d) / 128.0f) * __logf(10000.0f));
  float ang = p * inv;
  float c = __cosf(ang), s = __sinf(ang);
  size_t base = (size_t)i * 4096 + (size_t)h * 128 + d;
  float x1 = bf2f(Q[base]), x2 = bf2f(Q[base + 64]);
  Q[base]      = f2bf(x1 * c - x2 * s);
  Q[base + 64] = f2bf(x2 * c + x1 * s);
  float y1 = bf2f(Kk[base]), y2 = bf2f(Kk[base + 64]);
  Kk[base]      = f2bf(y1 * c - y2 * s);
  Kk[base + 64] = f2bf(y2 * c + y1 * s);
}

// ---------------------------------------------------------------------------
// V transpose (LDS tiled): Vt[c][j] = Vb[j][c], c in [0,4096), j in [0,S)
// ---------------------------------------------------------------------------
__global__ void vtrans_kernel(const bf16* __restrict__ Vb,
                              bf16* __restrict__ Vt, int S) {
  __shared__ bf16 tile[32][33];
  const int c0 = blockIdx.x * 32;
  const int j0 = blockIdx.y * 32;
  const int tx = threadIdx.x & 31;
  const int ty = threadIdx.x >> 5;   // 0..7
#pragma unroll
  for (int r = ty; r < 32; r += 8)
    tile[r][tx] = Vb[(size_t)(j0 + r) * 4096 + (c0 + tx)];
  __syncthreads();
#pragma unroll
  for (int r = ty; r < 32; r += 8)
    Vt[(size_t)(c0 + r) * (size_t)S + (j0 + tx)] = tile[tx][r];
}

// ---------------------------------------------------------------------------
// Per-row top-k threshold (bitwise bisection on order-preserving u32 keys)
// + softmax over kept set.  One 256-thread block per query row.
// ---------------------------------------------------------------------------
__global__ void topk_softmax_kernel(const float* __restrict__ Sc,
                                    bf16* __restrict__ Wt,
                                    const int* __restrict__ kptr, int S) {
  const int row = blockIdx.x;
  const int tid = threadIdx.x;           // 256 threads
  const int PER = S >> 8;                // 8 elements / thread
  const float* srow = Sc + (size_t)row * (size_t)S;
  bf16* wrow = Wt + (size_t)row * (size_t)S;

  __shared__ float red[256];
  __shared__ int cnt;
  const int kk = *kptr;

  float    val[8];
  unsigned key[8];
  float mymax = -INFINITY;
#pragma unroll
  for (int e = 0; e < PER; ++e) {
    float v = srow[e * 256 + tid];
    val[e] = v;
    unsigned u = __float_as_uint(v);
    key[e] = (u & 0x80000000u) ? ~u : (u | 0x80000000u);  // orderable
    mymax = fmaxf(mymax, v);
  }
  red[tid] = mymax; __syncthreads();
  for (int s2 = 128; s2 > 0; s2 >>= 1) {
    if (tid < s2) red[tid] = fmaxf(red[tid], red[tid + s2]);
    __syncthreads();
  }
  const float m = red[0];
  __syncthreads();

  unsigned T = 0;                        // largest T with count(key>=T) >= k
  for (int b = 31; b >= 0; --b) {
    const unsigned cand = T | (1u << b);
    if (tid == 0) cnt = 0;
    __syncthreads();
    int c = 0;
#pragma unroll
    for (int e = 0; e < PER; ++e) c += (key[e] >= cand) ? 1 : 0;
    atomicAdd(&cnt, c);
    __syncthreads();
    if (cnt >= kk) T = cand;
    __syncthreads();
  }

  float mysum = 0.f;
#pragma unroll
  for (int e = 0; e < PER; ++e)
    if (key[e] >= T) mysum += __expf(val[e] - m);
  red[tid] = mysum; __syncthreads();
  for (int s2 = 128; s2 > 0; s2 >>= 1) {
    if (tid < s2) red[tid] += red[tid + s2];
    __syncthreads();
  }
  const float inv = 1.0f / red[0];
#pragma unroll
  for (int e = 0; e < PER; ++e) {
    float w = (key[e] >= T) ? __expf(val[e] - m) * inv : 0.0f;
    wrow[e * 256 + tid] = f2bf(w);
  }
}

// ---------------------------------------------------------------------------
extern "C" void kernel_launch(void* const* d_in, const int* in_sizes, int n_in,
                              void* d_out, int out_size, void* d_ws, size_t ws_size,
                              hipStream_t stream) {
  (void)in_sizes; (void)n_in; (void)out_size; (void)ws_size;
  const float* hs  = (const float*)d_in[0];
  const int*   pos = (const int*)d_in[2];
  const float* Wq  = (const float*)d_in[3];
  const float* Wk  = (const float*)d_in[4];
  const float* Wv  = (const float*)d_in[5];
  const float* Wo  = (const float*)d_in[6];
  const int*   kp  = (const int*)d_in[7];
  float* out = (float*)d_out;

  const int S = 2048, H = 4096;
  char* ws = (char*)d_ws;
  size_t off = 0;
  auto carve = [&](size_t bytes) -> void* {
    void* p = ws + off;
    off += (bytes + 255) & ~(size_t)255;
    return p;
  };
  bf16* Xb = (bf16*)carve((size_t)S * H * 2);         // hidden bf16
  bf16* Qb = (bf16*)carve((size_t)S * H * 2);
  bf16* Kb = (bf16*)carve((size_t)S * H * 2);
  bf16* Vb = (bf16*)carve((size_t)S * H * 2);
  bf16* Vt = (bf16*)carve((size_t)S * H * 2);         // V transposed [4096][S]
  bf16* AO = (bf16*)carve((size_t)S * H * 2);         // attn out (merged heads)
  float* Sc = (float*)carve((size_t)S * S * 4);       // per-head scores
  bf16* Wt = (bf16*)carve((size_t)S * S * 2);         // per-head softmax weights
  bf16* Wb = (bf16*)carve((size_t)H * H * 2);         // reusable bf16 weight slot

  const float scale = 0.08838834764831845f;           // 1/sqrt(128)
  const int NW = H * H;                               // weight elements

  // 1) X -> bf16
  cvt_f32_bf16_kernel<<<(S * H) / 1024, 256, 0, stream>>>(hs, Xb, S * H);

  // 2) Q,K,V projections: convert weight once, then pure-bf16 WMMA GEMM.
  {
    const int waves = (S / 32) * (H / 64);            // MT=2
    dim3 g(waves / 8), b(256);
    cvt_f32_bf16_kernel<<<NW / 1024, 256, 0, stream>>>(Wq, Wb, NW);
    wmma_gemm_kernel<0, 2><<<g, b, 0, stream>>>(Xb, H, Wb, H, Qb, H, S, H, H, 1.f);
    cvt_f32_bf16_kernel<<<NW / 1024, 256, 0, stream>>>(Wk, Wb, NW);
    wmma_gemm_kernel<0, 2><<<g, b, 0, stream>>>(Xb, H, Wb, H, Kb, H, S, H, H, 1.f);
    cvt_f32_bf16_kernel<<<NW / 1024, 256, 0, stream>>>(Wv, Wb, NW);
    wmma_gemm_kernel<0, 2><<<g, b, 0, stream>>>(Xb, H, Wb, H, Vb, H, S, H, H, 1.f);
  }

  // 3) RoPE in place on Q, K
  rope_kernel<<<(S * 32 * 64) / 256, 256, 0, stream>>>(Qb, Kb, pos, S);

  // 4) transpose V for the PV GEMM
  {
    dim3 g(H / 32, S / 32), b(256);
    vtrans_kernel<<<g, b, 0, stream>>>(Vb, Vt, S);
  }

  // 5) per-head attention
  for (int h = 0; h < 32; ++h) {
    // scores = Q_h K_h^T * scale + causal mask   (S x S, f32), MT=2
    {
      const int waves = (S / 32) * (S / 64);
      wmma_gemm_kernel<2, 2><<<waves / 8, 256, 0, stream>>>(
          Qb + h * 128, H, Kb + h * 128, H, Sc, S, S, S, 128, scale);
    }
    // top-k threshold + softmax -> dense bf16 weights row
    topk_softmax_kernel<<<S, 256, 0, stream>>>(Sc, Wt, kp, S);
    // out_h = Wt * V_h   (S x 128) into merged-head buffer; MT=1 for waves
    {
      const int waves = (S / 16) * (128 / 64);
      wmma_gemm_kernel<0, 1><<<waves / 8, 256, 0, stream>>>(
          Wt, S, Vt + (size_t)h * 128 * S, S, AO + h * 128, H, S, 128, S, 1.f);
    }
  }

  // 6) output projection: out = AO * Wo^T  (f32 result), MT=2
  {
    cvt_f32_bf16_kernel<<<NW / 1024, 256, 0, stream>>>(Wo, Wb, NW);
    const int waves = (S / 32) * (H / 64);
    wmma_gemm_kernel<1, 2><<<waves / 8, 256, 0, stream>>>(
        AO, H, Wb, H, out, H, S, H, H, 1.f);
  }
}
